// Decoder_46342697124095
// MI455X (gfx1250) — matrix operand order repaired and checked by reading, not compiled, not run on hardware
//
#include <hip/hip_runtime.h>
#include <hip/hip_bf16.h>
#include <math.h>

typedef __attribute__((ext_vector_type(16))) __bf16 bf16x16;
typedef __attribute__((ext_vector_type(8)))  __bf16 bf16x8;
typedef __attribute__((ext_vector_type(4)))  __bf16 bf16x4;
typedef __attribute__((ext_vector_type(8)))  float  f32x8;
typedef __attribute__((ext_vector_type(4)))  float  f32x4;

__device__ __forceinline__ float fast_sigmoid(float x) {
    return 1.0f / (1.0f + __expf(-x));
}
__device__ __forceinline__ float fast_tanh(float x) {
    float e = __expf(2.0f * x);
    return (e - 1.0f) / (e + 1.0f);
}

// One LSTM layer, full time scan, batch tile of BM rows per block.
// ACT: 0 = tanh (layer 1), 1 = sigmoid (layer 2).
// NTOUT: 1 = nontemporal output stores (final layer, write-once data).
// X: [B][T][DIN] f32, W: [DIN][4H] f32, U: [H][4H] f32, Bv: [4H] f32
// Hout: [B][T][H] f32
template<int DIN, int ACT, int NTOUT>
__global__ __launch_bounds__(256)
void lstm_scan(const float* __restrict__ X, const float* __restrict__ W,
               const float* __restrict__ U, const float* __restrict__ Bv,
               float* __restrict__ Hout)
{
    constexpr int T = 128, H = 64, NG = 256, BM = 32;
    constexpr int KTOT  = DIN + H;      // concat(x, h) depth
    constexpr int PITCH = KTOT + 8;     // LDS row pitch (halfs), 16B-aligned rows

    extern __shared__ __bf16 smem[];
    __bf16* Wt   = smem;                // [NG][PITCH]  Wt[n][k] = concat(W;U)[k][n]
    __bf16* Abuf = smem + NG * PITCH;   // [BM][PITCH]  rows = [x_t | h]

    const int tid = threadIdx.x;
    const int b0  = blockIdx.x * BM;

    // Stage transposed bf16 weights (one-time per block).
    for (int idx = tid; idx < KTOT * NG; idx += 256) {
        int k = idx >> 8;              // 0..KTOT-1
        int n = idx & (NG - 1);        // 0..255
        float w = (k < DIN) ? W[k * NG + n] : U[(k - DIN) * NG + n];
        Wt[n * PITCH + k] = (__bf16)w;
    }
    // Zero the h region of the A buffer (h(-1) = 0).
    for (int idx = tid; idx < BM * H; idx += 256) {
        int r = idx >> 6, c = idx & (H - 1);
        Abuf[r * PITCH + DIN + c] = (__bf16)0.0f;
    }

    const int wave  = tid >> 5;         // 0..7
    const int lane  = tid & 31;
    const int mtile = wave >> 2;        // 0..1  (batch row tile)
    const int htile = wave & 3;         // 0..3  (hidden-column tile)
    const int lmod  = lane & 15;
    const int lhi   = lane >> 4;        // 0/1
    const int hc    = htile * 16 + lmod;        // hidden column 0..63
    const int arow  = mtile * 16 + lmod;        // A row this lane fetches

    float bias[4];
#pragma unroll
    for (int g = 0; g < 4; ++g) bias[g] = Bv[g * H + hc];

    float cst[8];                        // cell state, rows M = lhi*8 + r
#pragma unroll
    for (int r = 0; r < 8; ++r) cst[r] = 0.0f;

    constexpr int D4 = DIN / 4;          // float4s per x row
    for (int t = 0; t < T; ++t) {
        // Stage x_t tile (BM x DIN) into LDS as bf16 (vectorized, NT loads:
        // every element is read exactly once; keep the 192MB L2 for h state).
        for (int idx = tid; idx < BM * D4; idx += 256) {
            int r = idx / D4, d4 = idx % D4;
            const f32x4* xp = (const f32x4*)
                &X[((size_t)(b0 + r) * T + t) * DIN + 4 * d4];
            f32x4 xv = __builtin_nontemporal_load(xp);
            bf16x4 hv4;
            hv4.x = (__bf16)xv.x; hv4.y = (__bf16)xv.y;
            hv4.z = (__bf16)xv.z; hv4.w = (__bf16)xv.w;
            *(bf16x4*)&Abuf[r * PITCH + 4 * d4] = hv4;
        }
        __syncthreads();                 // x_t staged + h(t-1) visible

        // Zero-init accumulators: first WMMA of the K loop lowers its C
        // operand to the free inline-0; bias is folded into the epilogue.
        f32x8 acc[4] = {};

#pragma unroll
        for (int kk = 0; kk < KTOT; kk += 32) {
            // A fragment: 16x32 bf16, per ISA layout
            // lane<16: halves0-7 -> K kk..kk+7,  halves8-15 -> K kk+16..kk+23
            // lane>=16: +8 on both chunks
            union { bf16x16 v; bf16x8 h[2]; } au;
            const __bf16* ap = &Abuf[arow * PITCH + kk + (lhi ? 8 : 0)];
            au.h[0] = *(const bf16x8*)ap;
            au.h[1] = *(const bf16x8*)(ap + 16);
#pragma unroll
            for (int g = 0; g < 4; ++g) {
                // B fragment: 32x16 bf16, lane column N = g*64+hc,
                // K contiguous along the transposed-weight LDS row.
                union { bf16x16 v; bf16x8 h[2]; } bu;
                const __bf16* bp = &Wt[(g * H + hc) * PITCH + kk + (lhi ? 8 : 0)];
                bu.h[0] = *(const bf16x8*)bp;
                bu.h[1] = *(const bf16x8*)(bp + 16);
                acc[g] = __builtin_amdgcn_wmma_f32_16x16x32_bf16(
                    false, au.v, false, bu.v, (short)0, acc[g], false, false);
            }
        }
        __syncthreads();                 // all A-buffer reads done before overwrite

        // Gate combine (per lane: col hc, rows M = mtile*16 + lhi*8 + r)
#pragma unroll
        for (int r = 0; r < 8; ++r) {
            float iv = fast_sigmoid(acc[0][r] + bias[0]);
            float fv = fast_sigmoid(acc[1][r] + bias[1]);
            float zg = acc[2][r] + bias[2];
            float gv = (ACT == 0) ? fast_tanh(zg) : fast_sigmoid(zg);
            float ov = fast_sigmoid(acc[3][r] + bias[3]);
            float cn = fv * cst[r] + iv * gv;
            cst[r] = cn;
            float hv = ov * ((ACT == 0) ? fast_tanh(cn) : fast_sigmoid(cn));
            int m = mtile * 16 + lhi * 8 + r;
            Abuf[m * PITCH + DIN + hc] = (__bf16)hv;                 // recurrence
            float* op = &Hout[((size_t)(b0 + m) * T + t) * H + hc];
            if (NTOUT) __builtin_nontemporal_store(hv, op);          // write-once
            else       *op = hv;                                     // stays in L2
        }
    }
}

extern "C" void kernel_launch(void* const* d_in, const int* in_sizes, int n_in,
                              void* d_out, int out_size, void* d_ws, size_t ws_size,
                              hipStream_t stream) {
    const float* x  = (const float*)d_in[0];
    const float* W1 = (const float*)d_in[1];
    const float* U1 = (const float*)d_in[2];
    const float* b1 = (const float*)d_in[3];
    const float* W2 = (const float*)d_in[4];
    const float* U2 = (const float*)d_in[5];
    const float* b2 = (const float*)d_in[6];
    float* out = (float*)d_out;
    float* h1  = (float*)d_ws;          // [2048][128][64] f32 = 64 MB scratch

    constexpr int Bsz = 2048, BM = 32;
    constexpr size_t sh1 = (size_t)(256 + 32) * (128 + 64 + 8) * sizeof(__bf16);
    constexpr size_t sh2 = (size_t)(256 + 32) * ( 64 + 64 + 8) * sizeof(__bf16);

    (void)in_sizes; (void)n_in; (void)out_size; (void)ws_size;

    (void)hipFuncSetAttribute(reinterpret_cast<const void*>(&lstm_scan<128, 0, 0>),
                              hipFuncAttributeMaxDynamicSharedMemorySize, (int)sh1);
    (void)hipFuncSetAttribute(reinterpret_cast<const void*>(&lstm_scan<64, 1, 1>),
                              hipFuncAttributeMaxDynamicSharedMemorySize, (int)sh2);

    lstm_scan<128, 0, 0><<<Bsz / BM, 256, sh1, stream>>>(x,  W1, U1, b1, h1);
    lstm_scan< 64, 1, 1><<<Bsz / BM, 256, sh2, stream>>>(h1, W2, U2, b2, out);
}
